// Attention_66254165508472
// MI455X (gfx1250) — matrix operand order repaired and checked by reading, not compiled, not run on hardware
//
#include <hip/hip_runtime.h>

typedef __bf16 bf16_t;
typedef __bf16 v16bf __attribute__((ext_vector_type(16)));
typedef float  v8f   __attribute__((ext_vector_type(8)));

#define SEQ_L   2048
#define DMODEL  1024
#define NHEADS  8
#define HEADD   128
#define NBATCH  2
#define MTOT    (NBATCH * SEQ_L)   // 4096

static __device__ __forceinline__ int iclamp(int x, int lo, int hi) {
  return x < lo ? lo : (x > hi ? hi : x);
}

// ---- CDNA5 async global->LDS copy (ASYNCcnt-tracked, bypasses VGPRs) ----
// CDNA5 ISA: each lane copies 16 bytes from its global address to its LDS address.
static __device__ __forceinline__ void async_copy_b128(unsigned lds_addr,
                                                       const void* gaddr) {
  asm volatile("global_load_async_to_lds_b128 %0, %1, off"
               :
               : "v"(lds_addr), "v"(gaddr)
               : "memory");
}

template <int N>
static __device__ __forceinline__ void s_wait_async() {
#if __has_builtin(__builtin_amdgcn_s_wait_asynccnt)
  __builtin_amdgcn_s_wait_asynccnt(N);
#else
  asm volatile("s_wait_asynccnt %0" ::"n"(N) : "memory");
#endif
}

// ---------------- fp32 -> bf16 bulk convert (vectorized) ----------------
__global__ __launch_bounds__(256) void cvt_f32_bf16(const float* __restrict__ src,
                                                    bf16_t* __restrict__ dst, int n4) {
  int i = blockIdx.x * 256 + threadIdx.x;
  if (i >= n4) return;
  const float4 f = ((const float4*)src)[i];
  union { bf16_t h[4]; uint2 u; } pk;
  pk.h[0] = (bf16_t)f.x; pk.h[1] = (bf16_t)f.y;
  pk.h[2] = (bf16_t)f.z; pk.h[3] = (bf16_t)f.w;
  ((uint2*)dst)[i] = pk.u;
}

// ---------------- NT GEMM: C[m,n] = sum_k A[m,k] * W[n,k] ----------------
// A: [M x 1024] bf16 row-major, W: [1024 x 1024] bf16 row-major (N x K).
// Async double-buffered LDS staging; 8 waves, each owns a 64x32 C tile.
// OUT_MODE 0: store bf16 transposed-per-head V_t[b][h][d][q]
// OUT_MODE 1: store fp32 row-major [M x 1024]
template <int OUT_MODE>
__global__ __launch_bounds__(256) void gemm_nt_bf16(
    const bf16_t* __restrict__ A, const bf16_t* __restrict__ W,
    void* __restrict__ Cout) {
  constexpr int K = DMODEL;
  constexpr int NCHUNK = K / 32;   // 32
  __shared__ bf16_t As[2][128][40];   // +8 pad: 80B row stride, 16B-aligned runs
  __shared__ bf16_t Bs[2][128][40];

  const int tid   = threadIdx.x;
  const int wave  = tid >> 5;
  const int lane  = tid & 31;
  const int lrow  = lane & 15;
  const int lhalf = lane >> 4;
  const int wm    = (wave & 1) * 64;   // 2 waves over M (128)
  const int wn    = (wave >> 1) * 32;  // 4 waves over N (128)
  const int m_blk = blockIdx.y * 128;
  const int n_blk = blockIdx.x * 128;

  v8f acc[4][2] = {};

  const int srow = tid >> 2;        // staging: 4 threads per 64B row segment
  const int scol = (tid & 3) * 8;   // bf16 element offset (16B chunks)

  const bf16_t* ga = A + (size_t)(m_blk + srow) * K + scol;
  const bf16_t* gw = W + (size_t)(n_blk + srow) * K + scol;

  // issue 4 async 16B copies per thread for one 128x32 A tile + B tile
  auto stage = [&](int k0, int buf) {
#pragma unroll
    for (int i = 0; i < 2; ++i) {
      const int r = srow + i * 64;
      async_copy_b128((unsigned)(uintptr_t)&As[buf][r][scol],
                      ga + (size_t)i * 64 * K + k0);
      async_copy_b128((unsigned)(uintptr_t)&Bs[buf][r][scol],
                      gw + (size_t)i * 64 * K + k0);
    }
  };

  stage(0, 0);                      // prologue: chunk 0 -> buffer 0
  for (int t = 0; t < NCHUNK; ++t) {
    const int buf = t & 1;
    if (t + 1 < NCHUNK) {
      stage((t + 1) * 32, buf ^ 1); // prefetch next chunk into other buffer
      s_wait_async<4>();            // in-order: <=4 outstanding => chunk t done
    } else {
      s_wait_async<0>();
    }
    __syncthreads();

    union { v16bf v; uint4 u[2]; } af[4], bfr[2];
#pragma unroll
    for (int mf = 0; mf < 4; ++mf) {
      const int r = wm + mf * 16 + lrow;
      const int c = lhalf * 8;                     // K runs [c,c+7],[c+16,c+23]
      af[mf].u[0] = *(const uint4*)&As[buf][r][c];
      af[mf].u[1] = *(const uint4*)&As[buf][r][c + 16];
    }
#pragma unroll
    for (int nf = 0; nf < 2; ++nf) {
      const int r = wn + nf * 16 + lrow;
      const int c = lhalf * 8;                     // K runs [c,c+7],[c+16,c+23]
      bfr[nf].u[0] = *(const uint4*)&Bs[buf][r][c];
      bfr[nf].u[1] = *(const uint4*)&Bs[buf][r][c + 16];
    }
#pragma unroll
    for (int mf = 0; mf < 4; ++mf)
#pragma unroll
      for (int nf = 0; nf < 2; ++nf)
        acc[mf][nf] = __builtin_amdgcn_wmma_f32_16x16x32_bf16(
            false, af[mf].v, false, bfr[nf].v, (short)0, acc[mf][nf], false, false);
    __syncthreads();                // safe to overwrite this buffer next iter
  }

  if (OUT_MODE == 0) {
    bf16_t* Vt = (bf16_t*)Cout;
#pragma unroll
    for (int mf = 0; mf < 4; ++mf)
#pragma unroll
      for (int nf = 0; nf < 2; ++nf) {
        const int col = n_blk + wn + nf * 16 + lrow;       // model-D index
        const int h = col >> 7, d = col & (HEADD - 1);
        const int m = m_blk + wm + mf * 16 + lhalf * 8;    // 8 consecutive rows
        const int b = m >> 11, q = m & (SEQ_L - 1);
        union { uint4 u; bf16_t h8[8]; } pk;
#pragma unroll
        for (int e = 0; e < 8; ++e) pk.h8[e] = (bf16_t)acc[mf][nf][e];
        *(uint4*)&Vt[(((size_t)b * NHEADS + h) * HEADD + d) * SEQ_L + q] = pk.u;
      }
  } else {
    float* Co = (float*)Cout;
#pragma unroll
    for (int mf = 0; mf < 4; ++mf)
#pragma unroll
      for (int nf = 0; nf < 2; ++nf) {
        const int n  = n_blk + wn + nf * 16 + lrow;
        const int m0 = m_blk + wm + mf * 16 + lhalf * 8;
#pragma unroll
        for (int e = 0; e < 8; ++e)
          Co[(size_t)(m0 + e) * DMODEL + n] = acc[mf][nf][e];
      }
  }
}

// ---------------- banded Gaussian attention ----------------
// attended[b,h,q,0:128] = sum_k N(k-q-off_h; std_h) * V[b,h,k,0:128]
// One wave per 16-row q-tile; k window [q0-64, q0+95] in 5 chunks of 32
// (Gaussian < 1e-12 outside the band for std<=8).
__global__ __launch_bounds__(256) void attn_gauss(const bf16_t* __restrict__ Vt,
                                                  bf16_t* __restrict__ Att) {
  const int lane  = threadIdx.x & 31;
  const int wave  = threadIdx.x >> 5;
  const int lrow  = lane & 15;
  const int lhalf = lane >> 4;

  const int flat = blockIdx.x * 8 + wave;   // (b,h,qtile) flattened, 0..2047
  const int qt = flat & 127;
  const int h  = (flat >> 7) & 7;
  const int b  = flat >> 10;
  const int q0 = qt * 16;

  const float stdv    = (float)(1 << (h & 3));      // 1,2,4,8,1,2,4,8
  const float offv    = -stdv;                      // -1,-2,-4,-8,...
  const float inv_std = 1.0f / stdv;
  const float invn    = 0.39894228040143267794f * inv_std;  // 1/(std*sqrt(2pi))

  const int q = q0 + lrow;                          // this lane's query row

  v8f acc[8] = {};
  const bf16_t* vbase = Vt + ((size_t)b * NHEADS + h) * HEADD * SEQ_L;

  for (int t = 0; t < 5; ++t) {
    const int k0 = q0 - 64 + t * 32;

    // (a) issue ALL 16 B-loads for this k-chunk first (one clause, one wait);
    //     fully-OOB 16-runs are clamped (their probs are zero anyway)
    const int kst = iclamp(k0 + 16 * lhalf, 0, SEQ_L - 16);
    union { v16bf v; uint4 u[2]; } bfr[8];
#pragma unroll
    for (int nf = 0; nf < 8; ++nf) {
      const uint4* p = (const uint4*)(vbase + (size_t)(nf * 16 + lrow) * SEQ_L + kst);
      bfr[nf].u[0] = p[0];
      bfr[nf].u[1] = p[1];
    }

    // (b) generate probs A-fragment in-register (v_exp_f32 hides load latency)
    union { v16bf v; } af;
#pragma unroll
    for (int e = 0; e < 16; ++e) {
      const int kk = e + 16 * lhalf;                    // contiguous K mapping (matches B)
      const int k  = k0 + kk;
      const float s = ((float)(k - q) - offv) * inv_std;
      float p = invn * __expf(-0.5f * s * s);
      p = (k >= 0 && k < SEQ_L) ? p : 0.0f;             // honor sequence bounds
      af.v[e] = (bf16_t)p;
    }

    // (c) 8 back-to-back WMMAs
#pragma unroll
    for (int nf = 0; nf < 8; ++nf)
      acc[nf] = __builtin_amdgcn_wmma_f32_16x16x32_bf16(
          false, af.v, false, bfr[nf].v, (short)0, acc[nf], false, false);
  }

  // store attended as bf16 [b, q, h*128+d] (A-operand layout for GEMM2)
  const int qb = q0 + lhalf * 8;
#pragma unroll
  for (int nf = 0; nf < 8; ++nf) {
    const int col = h * HEADD + nf * 16 + lrow;
#pragma unroll
    for (int e = 0; e < 8; ++e)
      Att[((size_t)b * SEQ_L + qb + e) * DMODEL + col] = (bf16_t)acc[nf][e];
  }
}

// ---------------- launcher ----------------
extern "C" void kernel_launch(void* const* d_in, const int* in_sizes, int n_in,
                              void* d_out, int out_size, void* d_ws, size_t ws_size,
                              hipStream_t stream) {
  (void)in_sizes; (void)n_in; (void)out_size; (void)ws_size;
  const float* values = (const float*)d_in[0];
  const float* Win    = (const float*)d_in[1];
  const float* Wout   = (const float*)d_in[2];
  float* out = (float*)d_out;

  char* ws = (char*)d_ws;
  bf16_t* val_bf  = (bf16_t*)(ws);                        //  8 MB: values bf16
  bf16_t* win_bf  = (bf16_t*)(ws + ((size_t)8  << 20));   //  2 MB
  bf16_t* wout_bf = (bf16_t*)(ws + ((size_t)10 << 20));   //  2 MB
  bf16_t* vt      = (bf16_t*)(ws + ((size_t)12 << 20));   //  8 MB: V_t[b][h][d][q]
  bf16_t* att     = (bf16_t*)(ws + ((size_t)20 << 20));   //  8 MB: attended bf16

  cvt_f32_bf16<<<(MTOT * DMODEL / 4) / 256, 256, 0, stream>>>(values, val_bf, MTOT * DMODEL / 4);
  cvt_f32_bf16<<<(DMODEL * DMODEL / 4) / 256, 256, 0, stream>>>(Win,  win_bf,  DMODEL * DMODEL / 4);
  cvt_f32_bf16<<<(DMODEL * DMODEL / 4) / 256, 256, 0, stream>>>(Wout, wout_bf, DMODEL * DMODEL / 4);

  dim3 gg(DMODEL / 128, MTOT / 128);   // (8, 32)
  gemm_nt_bf16<0><<<gg, 256, 0, stream>>>(val_bf, win_bf, (void*)vt);
  attn_gauss<<<NBATCH * NHEADS * (SEQ_L / 16) / 8, 256, 0, stream>>>(vt, att);
  gemm_nt_bf16<1><<<gg, 256, 0, stream>>>(att, wout_bf, (void*)out);
}
